// AffineCoupling_69698729279524
// MI455X (gfx1250) — matrix operand order repaired and checked by reading, not compiled, not run on hardware
//
#include <hip/hip_runtime.h>

typedef unsigned short u16;
typedef unsigned int   u32;
typedef __bf16 v16bf __attribute__((ext_vector_type(16)));
typedef float  v8f   __attribute__((ext_vector_type(8)));

union ABfrag { v16bf v; uint4 q[2]; };

__device__ __forceinline__ u16 f2bf(float f) {
    __bf16 h = (__bf16)f;
    return __builtin_bit_cast(u16, h);
}

#define S_     64
#define NSQ_   32
#define L_     2048
#define NCHA_  512
#define SEMB_  128
#define PADR   520   // LDS row stride in bf16 elements (16B-aligned, conflict-skewed)

// ---------------- prep kernels ----------------

__global__ void k_zero_logdet(float* logdet) {
    if (threadIdx.x < 64) logdet[threadIdx.x] = 0.0f;
}

// wdyn[s][c][k] = emb[s]·Wa[c*3+k] + ba ; bdyn[s][c] = emb[s]·Wb[c] + bb
__global__ __launch_bounds__(256) void k_dyn(const float* __restrict__ emb,
                                             const float* __restrict__ Wa, const float* __restrict__ ba,
                                             const float* __restrict__ Wb, const float* __restrict__ bb,
                                             float* __restrict__ wdyn, float* __restrict__ bdyn) {
    int t = blockIdx.x * 256 + threadIdx.x;        // 64*2048 threads
    int s = t >> 11, r = t & 2047;
    const float* e = emb + s * SEMB_;
    if (r < 1536) {
        float acc = ba[r];
        const float* wr = Wa + r * SEMB_;
        for (int k = 0; k < SEMB_; ++k) acc += e[k] * wr[k];
        wdyn[s * 1536 + r] = acc;
    } else {
        int rr = r - 1536;
        float acc = bb[rr];
        const float* wr = Wb + rr * SEMB_;
        for (int k = 0; k < SEMB_; ++k) acc += e[k] * wr[k];
        bdyn[s * NCHA_ + rr] = acc;
    }
}

// W1bf[o*512+c] = bf16(W1[o][c][0]) ; W2bf[k][j][c] = bf16(W2[j][c][k])
__global__ __launch_bounds__(256) void k_cvt(const float* __restrict__ W1, const float* __restrict__ W2,
                                             u16* __restrict__ W1bf, u16* __restrict__ W2bf) {
    int idx = blockIdx.x * 256 + threadIdx.x;      // 262144 + 49152
    if (idx < 262144) {
        W1bf[idx] = f2bf(W1[idx]);
    } else {
        int j2 = idx - 262144;
        int k = j2 / 16384, rem = j2 & 16383;
        int j = rem >> 9, c = rem & 511;
        W2bf[j2] = f2bf(W2[(j * 512 + c) * 3 + k]);
    }
}

// ---------------- main GEMM: Y = relu(W1 @ relu(depthwise(h1)+b)) ----------------
// grid: (32 l-tiles of 64, 4 m-blocks of 128, 64 samples), 256 threads (8 waves)
__global__ __launch_bounds__(256) void k_gemm1(const float* __restrict__ h,
                                               const float* __restrict__ wdyn,
                                               const float* __restrict__ bdyn,
                                               const u16* __restrict__ W1bf,
                                               const float* __restrict__ b1,
                                               u16* __restrict__ Ybf) {
    extern __shared__ u16 xlds[];                  // [64][PADR] bf16, X transposed [l][c]
    const int s  = blockIdx.z;
    const int l0 = blockIdx.x * 64;
    const int mb = blockIdx.y * 128;
    const int tid = threadIdx.x;

    // Phase A: dynamic depthwise conv + bias + relu -> LDS (transposed)
    for (int it = tid; it < 64 * 256; it += 256) {
        int l  = it & 63;
        int c0 = (it >> 6) << 1;                   // even channel
        int q  = c0 >> 5;                          // h1 source channel
        int gl = l0 + l;
        const float* hrow = h + (size_t)(s * NSQ_ + q) * L_;
        float hm = (gl > 0)        ? hrow[gl - 1] : 0.0f;
        float hc = hrow[gl];
        float hp = (gl < L_ - 1)   ? hrow[gl + 1] : 0.0f;
        const float* wp = wdyn + (size_t)(s * NCHA_ + c0) * 3;
        float x0 = wp[0] * hm + wp[1] * hc + wp[2] * hp + bdyn[s * NCHA_ + c0];
        float x1 = wp[3] * hm + wp[4] * hc + wp[5] * hp + bdyn[s * NCHA_ + c0 + 1];
        x0 = fmaxf(x0, 0.0f); x1 = fmaxf(x1, 0.0f);
        u32 pk = (u32)f2bf(x0) | ((u32)f2bf(x1) << 16);
        *(u32*)&xlds[l * PADR + c0] = pk;
    }
    __syncthreads();

    // Phase B: WMMA GEMM. wave -> 32 rows x 32 cols (2x2 tiles of 16x16)
    const int w = tid >> 5, lane = tid & 31;
    const int laneq = lane & 15, half = lane >> 4;
    const int m0 = mb + (w & 3) * 32;              // global output row base
    const int n0 = (w >> 2) * 32;                  // local col base within l-tile

    v8f acc[2][2] = {};
    for (int kc = 0; kc < 16; ++kc) {
        const int ck = kc * 32;
        ABfrag a[2], b[2];
        #pragma unroll
        for (int mt = 0; mt < 2; ++mt) {           // A: 16x32 bf16, K = 16*half+2v+j
            int m = m0 + mt * 16 + laneq;
            a[mt].q[0] = *(const uint4*)(W1bf + m * 512 + ck + 16 * half);
            a[mt].q[1] = *(const uint4*)(W1bf + m * 512 + ck + 16 * half + 8);
        }
        #pragma unroll
        for (int nt = 0; nt < 2; ++nt) {           // B: 32x16 bf16, K = 16*half+2v+j
            int l = n0 + nt * 16 + laneq;
            const uint4* bp = (const uint4*)&xlds[l * PADR + ck + 16 * half];
            b[nt].q[0] = bp[0];
            b[nt].q[1] = bp[1];
        }
        #pragma unroll
        for (int mt = 0; mt < 2; ++mt)
            #pragma unroll
            for (int nt = 0; nt < 2; ++nt)
                acc[mt][nt] = __builtin_amdgcn_wmma_f32_16x16x32_bf16(
                    false, a[mt].v, false, b[nt].v, (short)0, acc[mt][nt], false, false);
    }

    // Epilogue: +b1, relu, store bf16 Y[s][m][l]
    #pragma unroll
    for (int mt = 0; mt < 2; ++mt)
        #pragma unroll
        for (int nt = 0; nt < 2; ++nt) {
            int l = l0 + n0 + nt * 16 + laneq;
            #pragma unroll
            for (int v = 0; v < 8; ++v) {
                int m = m0 + mt * 16 + v + 8 * half;
                float y = fmaxf(acc[mt][nt][v] + b1[m], 0.0f);
                Ybf[(size_t)(s * NCHA_ + m) * L_ + l] = f2bf(y);
            }
        }
}

// ---------------- conv2 (3-tap, 512->32) + sigmoid/affine/logdet ----------------
// grid: (16 l-tiles of 128, 1, 64 samples), 256 threads (8 waves)
__global__ __launch_bounds__(256) void k_conv2(const float* __restrict__ h,
                                               const u16* __restrict__ Ybf,
                                               const u16* __restrict__ W2bf,
                                               const float* __restrict__ b2,
                                               float* __restrict__ out,
                                               float* __restrict__ logdet) {
    extern __shared__ u16 ylds[];                  // [130][PADR] : rows = l0-1 .. l0+128
    const int s  = blockIdx.z;
    const int l0 = blockIdx.x * 128;
    const int tid = threadIdx.x;
    const int w = tid >> 5, lane = tid & 31;

    // stage Y tile (with halo), transposed to [l][c]
    for (int c = w; c < NCHA_; c += 8) {
        const u16* yr = Ybf + (size_t)(s * NCHA_ + c) * L_;
        for (int r = lane; r < 130; r += 32) {
            int gl = l0 - 1 + r;
            ylds[r * PADR + c] = (gl >= 0 && gl < L_) ? yr[gl] : (u16)0;
        }
    }
    __syncthreads();

    // GEMM: z[32 x 16cols] = sum_k W2[:,:,k] @ Y[:, l+k-1]; wave w owns cols w*16..+16
    const int laneq = lane & 15, half = lane >> 4;
    v8f acc[2] = {};
    for (int k = 0; k < 3; ++k) {
        const u16* W2p = W2bf + k * 32 * 512;
        for (int kc = 0; kc < 16; ++kc) {
            const int ck = kc * 32;
            ABfrag b;
            int r = w * 16 + laneq + k;            // LDS row for col l_local, tap k
            const uint4* bp = (const uint4*)&ylds[r * PADR + ck + 16 * half];
            b.q[0] = bp[0];
            b.q[1] = bp[1];
            #pragma unroll
            for (int mt = 0; mt < 2; ++mt) {
                ABfrag a;
                int j = mt * 16 + laneq;
                a.q[0] = *(const uint4*)(W2p + j * 512 + ck + 16 * half);
                a.q[1] = *(const uint4*)(W2p + j * 512 + ck + 16 * half + 8);
                acc[mt] = __builtin_amdgcn_wmma_f32_16x16x32_bf16(
                    false, a.v, false, b.v, (short)0, acc[mt], false, false);
            }
        }
    }

    // epilogue: s = sigmoid(z_s + 2) + 1e-7 ; h2' = s*(h2 + z_m) ; logdet += log(s)
    float lsum = 0.0f;
    const int l = l0 + w * 16 + laneq;
    #pragma unroll
    for (int v = 0; v < 8; ++v) {
        int j = v + 8 * half;                      // 0..15
        float zs = acc[0][v] + b2[j];
        float zm = acc[1][v] + b2[16 + j];
        float sv = 1.0f / (1.0f + __expf(-(zs + 2.0f))) + 1e-7f;
        size_t oidx = (size_t)(s * NSQ_ + 16 + j) * L_ + l;
        float h2v = h[oidx];
        out[oidx] = sv * (h2v + zm);
        lsum += __logf(sv);
    }
    atomicAdd(&logdet[s], lsum);

    // copy h1 passthrough (channels 0..15)
    for (int it = tid; it < 16 * 128; it += 256) {
        int j = it >> 7, ll = it & 127;
        size_t idx = (size_t)(s * NSQ_ + j) * L_ + l0 + ll;
        out[idx] = h[idx];
    }
}

// ---------------- host ----------------

extern "C" void kernel_launch(void* const* d_in, const int* in_sizes, int n_in,
                              void* d_out, int out_size, void* d_ws, size_t ws_size,
                              hipStream_t stream) {
    const float* h   = (const float*)d_in[0];
    const float* emb = (const float*)d_in[1];
    const float* Wa  = (const float*)d_in[2];
    const float* ba  = (const float*)d_in[3];
    const float* Wb  = (const float*)d_in[4];
    const float* bb  = (const float*)d_in[5];
    const float* W1  = (const float*)d_in[6];
    const float* b1  = (const float*)d_in[7];
    const float* W2  = (const float*)d_in[8];
    const float* b2  = (const float*)d_in[9];

    float* out    = (float*)d_out;
    float* logdet = out + (size_t)S_ * NSQ_ * L_;  // 4,194,304 .. +64

    char* ws = (char*)d_ws;
    float* wdyn = (float*)(ws + 0);                         //  64*1536*4 = 393,216
    float* bdyn = (float*)(ws + 393216);                    //  64*512*4  = 131,072
    u16*   W1bf = (u16*)(ws + 524288);                      //  512*512*2 = 524,288
    u16*   W2bf = (u16*)(ws + 1048576);                     //  3*32*512*2 = 98,304
    u16*   Ybf  = (u16*)(ws + 1146880);                     //  64*512*2048*2 = 134,217,728

    const int lds1 = 64  * PADR * sizeof(u16);              //  66,560 B
    const int lds2 = 130 * PADR * sizeof(u16);              // 135,200 B (gfx1250 WGP: 320 KB)
    (void)hipFuncSetAttribute((const void*)k_gemm1, hipFuncAttributeMaxDynamicSharedMemorySize, lds1);
    (void)hipFuncSetAttribute((const void*)k_conv2, hipFuncAttributeMaxDynamicSharedMemorySize, lds2);

    k_zero_logdet<<<1, 64, 0, stream>>>(logdet);
    k_dyn<<<(S_ * 2048) / 256, 256, 0, stream>>>(emb, Wa, ba, Wb, bb, wdyn, bdyn);
    k_cvt<<<(262144 + 49152) / 256, 256, 0, stream>>>(W1, W2, W1bf, W2bf);
    k_gemm1<<<dim3(32, 4, S_), 256, lds1, stream>>>(h, wdyn, bdyn, W1bf, b1, Ybf);
    k_conv2<<<dim3(16, 1, S_), 256, lds2, stream>>>(h, Ybf, W2bf, b2, out, logdet);
}
